// SpcConvWrapper_76948634075584
// MI455X (gfx1250) — compile-verified, matrix operand order checked by reading, not run on hardware
//
#include <hip/hip_runtime.h>

typedef __attribute__((ext_vector_type(16))) __bf16 v16bf;
typedef __attribute__((ext_vector_type(8)))  __bf16 v8bf;
typedef __attribute__((ext_vector_type(8)))  float  v8f;

#define CIN   128
#define COUT  128
#define NTAPS 27
#define MT    64            // points (rows) per workgroup
#define LDA   (CIN + 8)     // bf16 element stride of LDS A tile (bank spread)
#define LDO   (COUT + 4)    // f32 element stride of LDS out tile
#define ABUF  (MT * LDA)

struct SmemT {
  union {
    unsigned short A[2][ABUF];  // double-buffered bf16 A tiles
    float o[MT * LDO];          // f32 epilogue staging (after tap loop)
  };
  int nbr[NTAPS][MT];           // neighbor row index per tap (-1 = missing)
};

// ---------- prep: int64 voxel codes (sorted by construction) ----------
__global__ void spc_codes_kernel(const int* __restrict__ pts, const int* __restrict__ level,
                                 long long* __restrict__ codes, int n) {
  int i = blockIdx.x * blockDim.x + threadIdx.x;
  if (i >= n) return;
  long long r = 1ll << level[0];
  long long p0 = pts[i * 3 + 0], p1 = pts[i * 3 + 1], p2 = pts[i * 3 + 2];
  codes[i] = (p0 * r + p1) * r + p2;
}

// ---------- prep: weight [27][cin][cout] f32 -> [27][cout][cin] bf16 ----------
__global__ void spc_wt_kernel(const float* __restrict__ w, unsigned short* __restrict__ wt) {
  int idx = blockIdx.x * blockDim.x + threadIdx.x;        // [k][cout][cin] linear
  if (idx >= NTAPS * CIN * COUT) return;
  int k   = idx / (CIN * COUT);
  int rem = idx % (CIN * COUT);
  int nn  = rem / CIN;
  int kk  = rem % CIN;
  wt[idx] = __builtin_bit_cast(unsigned short, (__bf16)w[(k * CIN + kk) * COUT + nn]);
}

// ---------- prep: x f32 -> bf16 (L2-resident feature table, 102 MB) ----------
__global__ void spc_xb_kernel(const float* __restrict__ x, unsigned short* __restrict__ xb,
                              size_t total) {
  size_t i = ((size_t)blockIdx.x * blockDim.x + threadIdx.x) * 8;
  if (i >= total) return;
  const float4* s = (const float4*)(x + i);
  float4 a = s[0], b = s[1];
  ushort4 lo, hi;
  lo.x = __builtin_bit_cast(unsigned short, (__bf16)a.x);
  lo.y = __builtin_bit_cast(unsigned short, (__bf16)a.y);
  lo.z = __builtin_bit_cast(unsigned short, (__bf16)a.z);
  lo.w = __builtin_bit_cast(unsigned short, (__bf16)a.w);
  hi.x = __builtin_bit_cast(unsigned short, (__bf16)b.x);
  hi.y = __builtin_bit_cast(unsigned short, (__bf16)b.y);
  hi.z = __builtin_bit_cast(unsigned short, (__bf16)b.z);
  hi.w = __builtin_bit_cast(unsigned short, (__bf16)b.w);
  *(ushort4*)(xb + i)     = lo;
  *(ushort4*)(xb + i + 4) = hi;
}

// ---------- main: async/sync gather -> bf16 WMMA accumulate -> LN epilogue ----------
template <bool ASYNC>
__global__ __launch_bounds__(256)
void spc_conv_kernel(const int* __restrict__ pts, const int* __restrict__ level,
                     const float* __restrict__ x, const unsigned short* __restrict__ xb,
                     const int* __restrict__ kv,
                     const float* __restrict__ bias, const float* __restrict__ gamma,
                     const float* __restrict__ beta,
                     const long long* __restrict__ codes,
                     const unsigned short* __restrict__ wt,
                     float* __restrict__ out, int n) {
  __shared__ SmemT sm;

  const int t    = threadIdx.x;
  const int lane = t & 31;
  const int wv   = t >> 5;          // wave id = 16-column output tile
  const int half = lane >> 4;
  const int lrow = lane & 15;
  const int base = blockIdx.x * MT;
  const int r    = 1 << level[0];

  // ---- 1) all 27*64 neighbor binary searches, once, spread over 256 threads ----
  for (int i = t; i < NTAPS * MT; i += 256) {
    int k = i / MT, row = i % MT;
    int gr = base + row;
    int idx = -1;
    if (gr < n) {
      int q0 = pts[gr*3+0] + kv[k*3+0];
      int q1 = pts[gr*3+1] + kv[k*3+1];
      int q2 = pts[gr*3+2] + kv[k*3+2];
      if ((unsigned)q0 < (unsigned)r && (unsigned)q1 < (unsigned)r && (unsigned)q2 < (unsigned)r) {
        long long qc = (((long long)q0 * r) + q1) * r + q2;
        int lo = 0, hi = n;
        while (lo < hi) { int mid = (lo + hi) >> 1; if (codes[mid] < qc) lo = mid + 1; else hi = mid; }
        if (lo < n && codes[lo] == qc) idx = lo;
      }
    }
    sm.nbr[k][row] = idx;
  }
  __syncthreads();

  const int grow = t >> 2, gpart = t & 3;   // gather: 4 threads per row, 64B each

  // issue gather of tap k into buffer buf
  auto gather = [&](int k, int buf) {
    int idx = sm.nbr[k][grow];
    unsigned short* dst = &sm.A[buf][grow * LDA + gpart * 32];
    if (ASYNC) {
      if (idx >= 0) {
        unsigned long long ga = (unsigned long long)(size_t)(xb + (size_t)idx * CIN + gpart * 32);
        unsigned la = (unsigned)(size_t)dst;
        asm volatile("global_load_async_to_lds_b128 %0, %1, off"            :: "v"(la), "v"(ga) : "memory");
        asm volatile("global_load_async_to_lds_b128 %0, %1, off offset:16"  :: "v"(la), "v"(ga) : "memory");
        asm volatile("global_load_async_to_lds_b128 %0, %1, off offset:32"  :: "v"(la), "v"(ga) : "memory");
        asm volatile("global_load_async_to_lds_b128 %0, %1, off offset:48"  :: "v"(la), "v"(ga) : "memory");
      } else {
        uint4 z = {0u, 0u, 0u, 0u};
        #pragma unroll
        for (int u = 0; u < 4; ++u) *(uint4*)(dst + u * 8) = z;
      }
    } else {
      if (idx >= 0) {
        const float4* src = (const float4*)&x[(size_t)idx * CIN + gpart * 32];
        #pragma unroll
        for (int u = 0; u < 8; ++u) {
          float4 v = src[u];
          dst[u*4+0] = __builtin_bit_cast(unsigned short, (__bf16)v.x);
          dst[u*4+1] = __builtin_bit_cast(unsigned short, (__bf16)v.y);
          dst[u*4+2] = __builtin_bit_cast(unsigned short, (__bf16)v.z);
          dst[u*4+3] = __builtin_bit_cast(unsigned short, (__bf16)v.w);
        }
      } else {
        #pragma unroll
        for (int u = 0; u < 32; ++u) dst[u] = 0;
      }
    }
  };

  v8f acc[4] = {};

  auto mma = [&](int k, int buf) {
    const unsigned short* wbase = wt + ((size_t)k * COUT + (wv*16 + lrow)) * (size_t)CIN;
    const unsigned short* abase = &sm.A[buf][lrow * LDA + half * 8];
    #pragma unroll
    for (int kc = 0; kc < 4; ++kc) {
      const unsigned short* bp = wbase + kc*32 + half*8;
      v8bf blo = *(const v8bf*)bp;
      v8bf bhi = *(const v8bf*)(bp + 16);
      v16bf bfrag = __builtin_shufflevector(blo, bhi, 0,1,2,3,4,5,6,7,8,9,10,11,12,13,14,15);
      v16bf afrag[4];
      #pragma unroll
      for (int m = 0; m < 4; ++m) {          // hoist all 4 A-frag loads before WMMAs
        const unsigned short* ap = abase + (m * 16) * LDA + kc * 32;
        v8bf alo = *(const v8bf*)ap;
        v8bf ahi = *(const v8bf*)(ap + 16);
        afrag[m] = __builtin_shufflevector(alo, ahi, 0,1,2,3,4,5,6,7,8,9,10,11,12,13,14,15);
      }
      #pragma unroll
      for (int m = 0; m < 4; ++m)
        acc[m] = __builtin_amdgcn_wmma_f32_16x16x32_bf16(
            false, afrag[m], false, bfrag, (short)0, acc[m], false, false);
    }
  };

  // ---- 2) double-buffered tap loop: gather k+1 overlaps WMMA of k ----
  gather(0, 0);
  if (ASYNC) asm volatile("s_wait_asynccnt 0" ::: "memory");
  __syncthreads();
  for (int k = 0; k < NTAPS; ++k) {
    int cur = k & 1;
    if (k + 1 < NTAPS) gather(k + 1, cur ^ 1);
    mma(k, cur);
    if (ASYNC) asm volatile("s_wait_asynccnt 0" ::: "memory");
    __syncthreads();
  }

  // ---- 3) stage accumulators to LDS (overlays A buffers) ----
  {
    int col = wv*16 + lrow;
    #pragma unroll
    for (int m = 0; m < 4; ++m) {
      #pragma unroll
      for (int rr = 0; rr < 8; ++rr)
        sm.o[(m*16 + half*8 + rr) * LDO + col] = acc[m][rr];
    }
  }
  __syncthreads();

  // ---- 4) epilogue: bias + LeakyReLU(0.2) + LayerNorm over 128 channels ----
  {
    int row = t >> 2, part = t & 3;
    int gr  = base + row;
    float vals[32];
    float sum = 0.f, sumsq = 0.f;
    const float*  srow = &sm.o[row * LDO + part * 32];
    const float4* b4   = (const float4*)&bias[part * 32];
    #pragma unroll
    for (int u = 0; u < 8; ++u) {
      float4 bb = b4[u];
      #pragma unroll
      for (int c = 0; c < 4; ++c) {
        float v = srow[u*4 + c] + ((const float*)&bb)[c];
        v = (v >= 0.f) ? v : 0.2f * v;
        vals[u*4 + c] = v;
        sum += v; sumsq += v * v;
      }
    }
    sum += __shfl_xor(sum, 1, 32);  sumsq += __shfl_xor(sumsq, 1, 32);
    sum += __shfl_xor(sum, 2, 32);  sumsq += __shfl_xor(sumsq, 2, 32);
    float mean = sum * (1.0f / COUT);
    float var  = sumsq * (1.0f / COUT) - mean * mean;
    float rstd = rsqrtf(var + 1e-5f);
    if (gr < n) {
      float4*       op = (float4*)&out[(size_t)gr * COUT + part * 32];
      const float4* g4 = (const float4*)&gamma[part * 32];
      const float4* e4 = (const float4*)&beta[part * 32];
      #pragma unroll
      for (int u = 0; u < 8; ++u) {
        float4 gg = g4[u], ee = e4[u], o;
        o.x = (vals[u*4+0] - mean) * rstd * gg.x + ee.x;
        o.y = (vals[u*4+1] - mean) * rstd * gg.y + ee.y;
        o.z = (vals[u*4+2] - mean) * rstd * gg.z + ee.z;
        o.w = (vals[u*4+3] - mean) * rstd * gg.w + ee.w;
        op[u] = o;
      }
    }
  }
}

extern "C" void kernel_launch(void* const* d_in, const int* in_sizes, int n_in,
                              void* d_out, int out_size, void* d_ws, size_t ws_size,
                              hipStream_t stream) {
  const int*   pts   = (const int*)d_in[1];
  const int*   level = (const int*)d_in[2];
  const float* x     = (const float*)d_in[5];
  const int*   kv    = (const int*)d_in[6];
  const float* w     = (const float*)d_in[7];
  const float* bias  = (const float*)d_in[8];
  const float* gamma = (const float*)d_in[9];
  const float* beta  = (const float*)d_in[10];
  const int n = in_sizes[1] / 3;

  auto align256 = [](size_t v) { return (v + 255) & ~(size_t)255; };
  size_t codes_bytes = align256((size_t)n * 8);
  size_t wt_bytes    = align256((size_t)NTAPS * CIN * COUT * 2);
  size_t xb_bytes    = (size_t)n * CIN * 2;

  char* ws = (char*)d_ws;
  long long*      codes = (long long*)ws;
  unsigned short* wtb   = (unsigned short*)(ws + codes_bytes);
  unsigned short* xb    = (unsigned short*)(ws + codes_bytes + wt_bytes);
  bool use_async = ws_size >= codes_bytes + wt_bytes + xb_bytes;

  spc_codes_kernel<<<(n + 255) / 256, 256, 0, stream>>>(pts, level, codes, n);
  spc_wt_kernel<<<(NTAPS * CIN * COUT + 255) / 256, 256, 0, stream>>>(w, wtb);

  int blocks = (n + MT - 1) / MT;
  if (use_async) {
    size_t total = (size_t)n * CIN;
    spc_xb_kernel<<<(unsigned)((total / 8 + 255) / 256), 256, 0, stream>>>(x, xb, total);
    spc_conv_kernel<true><<<blocks, 256, 0, stream>>>(
        pts, level, x, xb, kv, bias, gamma, beta, codes, wtb, (float*)d_out, n);
  } else {
    spc_conv_kernel<false><<<blocks, 256, 0, stream>>>(
        pts, level, x, xb, kv, bias, gamma, beta, codes, wtb, (float*)d_out, n);
  }
}
